// UNetAttentionBlock_2757369004779
// MI455X (gfx1250) — compile-verified, matrix operand order checked by reading, not run on hardware
//
#include <hip/hip_runtime.h>

#define BATCH 4
#define CH    128
#define HWN   4096
#define NH    4
#define DH    32
#define TC    384   // 3*CH

typedef __attribute__((ext_vector_type(16))) _Float16 v16h;
typedef __attribute__((ext_vector_type(8)))  _Float16 v8h;
typedef __attribute__((ext_vector_type(8)))  float    v8f;

static __device__ __forceinline__ v16h cat8(v8h lo, v8h hi) {
  return __builtin_shufflevector(lo, hi, 0,1,2,3,4,5,6,7,8,9,10,11,12,13,14,15);
}
static __device__ __forceinline__ v8h ld8(const _Float16* p) { return *(const v8h*)p; }

// async copy 16B/lane from global to LDS (ASYNCcnt-tracked, no VGPR staging)
static __device__ __forceinline__ void async_cp16(unsigned ldsoff, const _Float16* g) {
  asm volatile("global_load_async_to_lds_b128 %0, %1, off"
               :: "v"(ldsoff), "v"((unsigned long long)(uintptr_t)g) : "memory");
}
static __device__ __forceinline__ void wait_async0() {
  asm volatile("s_wait_asynccnt 0" ::: "memory");
}
static __device__ __forceinline__ void wait_ds0() {
  asm volatile("s_wait_dscnt 0" ::: "memory");
}

// ---------------- kernel 1: per-batch mean / rstd ----------------
__global__ __launch_bounds__(256) void stats_kernel(const float* __restrict__ x,
                                                    float* __restrict__ stats) {
  int b = blockIdx.x;
  const float4* xb = (const float4*)(x + (size_t)b * CH * HWN);
  float s = 0.f, s2 = 0.f;
  for (int i = threadIdx.x; i < (CH * HWN) / 4; i += 256) {
    float4 v = xb[i];
    s  += v.x + v.y + v.z + v.w;
    s2 += v.x * v.x + v.y * v.y + v.z * v.z + v.w * v.w;
  }
  __shared__ float sh[512];
  sh[threadIdx.x] = s; sh[256 + threadIdx.x] = s2;
  __syncthreads();
  for (int st = 128; st > 0; st >>= 1) {
    if ((int)threadIdx.x < st) {
      sh[threadIdx.x]       += sh[threadIdx.x + st];
      sh[256 + threadIdx.x] += sh[256 + threadIdx.x + st];
    }
    __syncthreads();
  }
  if (threadIdx.x == 0) {
    const float inv = 1.0f / (float)(CH * HWN);
    float mean = sh[0] * inv;
    float var  = sh[256] * inv - mean * mean;
    stats[2 * b]     = mean;
    stats[2 * b + 1] = rsqrtf(var + 1e-5f);
  }
}

// ---------------- kernel 2: normalize + affine -> fp16 (b, hw, c) ----------------
__global__ __launch_bounds__(256) void norm_kernel(const float* __restrict__ x,
                                                   const float* __restrict__ gw,
                                                   const float* __restrict__ gb,
                                                   const float* __restrict__ stats,
                                                   _Float16* __restrict__ xnt) {
  size_t i = ((size_t)blockIdx.x * 256 + threadIdx.x) * 4;   // (b, c, n) order
  int b = (int)(i >> 19);           // CH*HWN = 2^19
  int c = (int)((i >> 12) & 127);   // HWN = 2^12
  int n = (int)(i & 4095);
  float mean = stats[2 * b], rstd = stats[2 * b + 1];
  float g  = gw[c] * rstd;
  float be = gb[c] - mean * g;
  float4 xv = *(const float4*)(x + i);
  _Float16* dst = xnt + ((size_t)b * HWN + n) * CH + c;
  dst[0 * CH] = (_Float16)(xv.x * g + be);
  dst[1 * CH] = (_Float16)(xv.y * g + be);
  dst[2 * CH] = (_Float16)(xv.z * g + be);
  dst[3 * CH] = (_Float16)(xv.w * g + be);
}

// ---------------- kernel 3: fp32 -> fp16 copy (weights) ----------------
__global__ __launch_bounds__(256) void f2h_kernel(const float* __restrict__ s,
                                                  _Float16* __restrict__ d, int n) {
  int i = blockIdx.x * 256 + threadIdx.x;
  if (i < n) d[i] = (_Float16)s[i];
}

// ---------------- kernel 4: QKV GEMM (WMMA) ----------------
// Y[b,o,n] = sum_c W[o,c] * Xn[b,n,c] + bias[o]
// q -> qT (bh, n, d) * scale ; k -> kT (bh, n, d) ; v -> vD (bh, d, n)
__global__ __launch_bounds__(256) void qkv_gemm(const _Float16* __restrict__ wq,
                                                const _Float16* __restrict__ xnt,
                                                const float* __restrict__ qkvb,
                                                _Float16* __restrict__ qT,
                                                _Float16* __restrict__ kT,
                                                _Float16* __restrict__ vD) {
  int wid  = blockIdx.x * 8 + (threadIdx.x >> 5);
  int lane = threadIdx.x & 31;
  int b  = wid / (24 * 64);
  int r0 = wid % (24 * 64);
  int mT = r0 / 64;          // 16-row tile of o (0..23)
  int nG = r0 % 64;          // 64-col group of n
  int row   = lane & 15;
  int off   = (lane >> 4) * 8;  // row offset for C/D, K offset for A/B
  int m0    = mT * 16;

  v16h a[4];
  const _Float16* wrow = wq + (size_t)(m0 + row) * CH;
#pragma unroll
  for (int kk = 0; kk < 4; ++kk)
    a[kk] = cat8(ld8(wrow + kk * 32 + off), ld8(wrow + kk * 32 + off + 16));

  float bias[8];
#pragma unroll
  for (int r = 0; r < 8; ++r) bias[r] = qkvb[m0 + r + off];

  int which = m0 >> 7;          // 0=q 1=k 2=v
  int h     = (m0 >> 5) & 3;
  int dbase = m0 & 31;          // 0 or 16
  size_t bh = (size_t)b * NH + h;

  for (int nt = 0; nt < 4; ++nt) {
    int n = nG * 64 + nt * 16 + row;
    const _Float16* bc = xnt + ((size_t)b * HWN + n) * CH;
    v8f acc = {};
#pragma unroll
    for (int kk = 0; kk < 4; ++kk) {
      v16h bt = cat8(ld8(bc + kk * 32 + off), ld8(bc + kk * 32 + off + 16));
      acc = __builtin_amdgcn_wmma_f32_16x16x32_f16(false, a[kk], false, bt,
                                                   (short)0, acc, false, false);
    }
    if (which == 2) {
      _Float16* vp = vD + bh * DH * HWN + n;
#pragma unroll
      for (int r = 0; r < 8; ++r)
        vp[(size_t)(dbase + r + off) * HWN] = (_Float16)(acc[r] + bias[r]);
    } else {
      float scl = (which == 0) ? 0.17677669529663687f : 1.0f; // dh^-0.5 folded into q
      v8h pk;
#pragma unroll
      for (int r = 0; r < 8; ++r) pk[r] = (_Float16)((acc[r] + bias[r]) * scl);
      _Float16* dst = ((which == 0) ? qT : kT) + (bh * HWN + n) * DH + dbase + off;
      *(v8h*)dst = pk;
    }
  }
}

// ---------------- kernel 5: flash attention (async-LDS staged K/V, 64-key blocks) ----
// per wave: 16 queries; block stages K/V cooperatively; out -> aoT (b,n,c), c = d*NH+h
__global__ __launch_bounds__(256) void flash_attn(const _Float16* __restrict__ qT,
                                                  const _Float16* __restrict__ kT,
                                                  const _Float16* __restrict__ vD,
                                                  _Float16* __restrict__ aoT) {
  __shared__ __align__(16) _Float16 kbuf[2][64 * DH];   // (key, d)  8 KB
  __shared__ __align__(16) _Float16 vbuf[2][DH * 64];   // (d, key)  8 KB
  __shared__ __align__(16) _Float16 pbuf[8][16 * 64];   // P per wave 16 KB
  int tid  = threadIdx.x;
  int wave = tid >> 5, lane = tid & 31;
  int bh    = blockIdx.x >> 5;
  int qTile = (blockIdx.x & 31) * 8 + wave;
  int row = lane & 15, off = (lane >> 4) * 8;
  int b = bh >> 2, h = bh & 3;

  const _Float16* kbase = kT + (size_t)bh * HWN * DH;
  const _Float16* vbase = vD + (size_t)bh * DH * HWN;

  // per-thread async-copy mapping: 16B of K + 16B of V per 64-key block
  const _Float16* kg = kbase + tid * 8;                                   // + kb0*DH
  const _Float16* vg = vbase + (size_t)(tid >> 3) * HWN + (tid & 7) * 8;  // + kb0
  unsigned kl[2], vl[2];
#pragma unroll
  for (int i = 0; i < 2; ++i) {
    kl[i] = (unsigned)(uintptr_t)&kbuf[i][tid * 8];
    vl[i] = (unsigned)(uintptr_t)&vbuf[i][(tid >> 3) * 64 + (tid & 7) * 8];
  }
  // prologue: block 0 -> buffer 0
  async_cp16(kl[0], kg);
  async_cp16(vl[0], vg);

  const _Float16* qrow = qT + ((size_t)bh * HWN + qTile * 16 + row) * DH;
  v16h aQ = cat8(ld8(qrow + off), ld8(qrow + off + 16));

  v8f o0 = {}, o1 = {};
  float rowM[8], rowL[8];
#pragma unroll
  for (int r = 0; r < 8; ++r) { rowM[r] = -1e30f; rowL[r] = 0.f; }
  _Float16* pw = &pbuf[wave][0];

  for (int kb = 0; kb < HWN / 64; ++kb) {
    int cur = kb & 1;
    wait_async0();          // this wave's copies for block kb have landed
    __syncthreads();        // everyone's copies landed; prev buffer fully consumed
    if (kb + 1 < HWN / 64) {
      int nb0 = (kb + 1) * 64;
      async_cp16(kl[cur ^ 1], kg + (size_t)nb0 * DH);
      async_cp16(vl[cur ^ 1], vg + nb0);
    }
    const _Float16* kl_lds = &kbuf[cur][0];
    const _Float16* vl_lds = &vbuf[cur][0];

    // S = Q^T K  : 4 tiles of 16 keys
    v8f s[4];
#pragma unroll
    for (int kt = 0; kt < 4; ++kt) {
      const _Float16* kr = kl_lds + (kt * 16 + row) * DH + off;
      v16h bK = cat8(*(const v8h*)kr, *(const v8h*)(kr + 16));
      v8f z = {};
      s[kt] = __builtin_amdgcn_wmma_f32_16x16x32_f16(false, aQ, false, bK,
                                                     (short)0, z, false, false);
    }
    // online softmax over 64 keys
#pragma unroll
    for (int r = 0; r < 8; ++r) {
      float mx = fmaxf(fmaxf(s[0][r], s[1][r]), fmaxf(s[2][r], s[3][r]));
#pragma unroll
      for (int msk = 1; msk < 16; msk <<= 1) mx = fmaxf(mx, __shfl_xor(mx, msk, 32));
      float newM = fmaxf(rowM[r], mx);
      float f = __expf(rowM[r] - newM);
      float p0 = __expf(s[0][r] - newM);
      float p1 = __expf(s[1][r] - newM);
      float p2 = __expf(s[2][r] - newM);
      float p3 = __expf(s[3][r] - newM);
      float rs = (p0 + p1) + (p2 + p3);
#pragma unroll
      for (int msk = 1; msk < 16; msk <<= 1) rs += __shfl_xor(rs, msk, 32);
      rowL[r] = rowL[r] * f + rs;
      rowM[r] = newM;
      o0[r] *= f; o1[r] *= f;
      pw[(r + off) * 64 +  0 + row] = (_Float16)p0;
      pw[(r + off) * 64 + 16 + row] = (_Float16)p1;
      pw[(r + off) * 64 + 32 + row] = (_Float16)p2;
      pw[(r + off) * 64 + 48 + row] = (_Float16)p3;
    }
    wait_ds0();             // wave-private P bounce (D-layout -> A-layout)
    // O += P * V^T : two 32-key chunks x two 16-wide d tiles
#pragma unroll
    for (int c = 0; c < 2; ++c) {
      const _Float16* pr = pw + row * 64 + c * 32 + off;
      v16h aP = cat8(*(const v8h*)pr, *(const v8h*)(pr + 16));
      const _Float16* v0 = vl_lds + row * 64 + c * 32 + off;
      const _Float16* v1 = vl_lds + (16 + row) * 64 + c * 32 + off;
      v16h bV0 = cat8(*(const v8h*)v0, *(const v8h*)(v0 + 16));
      v16h bV1 = cat8(*(const v8h*)v1, *(const v8h*)(v1 + 16));
      o0 = __builtin_amdgcn_wmma_f32_16x16x32_f16(false, aP, false, bV0, (short)0, o0, false, false);
      o1 = __builtin_amdgcn_wmma_f32_16x16x32_f16(false, aP, false, bV1, (short)0, o1, false, false);
    }
  }
#pragma unroll
  for (int r = 0; r < 8; ++r) {
    float inv = 1.0f / rowL[r];
    int n = qTile * 16 + r + off;
    _Float16* arow = aoT + ((size_t)b * HWN + n) * CH;   // c = d*NH + h (reference order)
    arow[row * 4 + h]        = (_Float16)(o0[r] * inv);  // d = row
    arow[(16 + row) * 4 + h] = (_Float16)(o1[r] * inv);  // d = 16 + row
  }
}

// ---------------- kernel 6: proj GEMM + bias + residual (WMMA) ----------------
__global__ __launch_bounds__(256) void proj_gemm(const _Float16* __restrict__ wp,
                                                 const _Float16* __restrict__ aoT,
                                                 const float* __restrict__ pb,
                                                 const float* __restrict__ x,
                                                 float* __restrict__ out) {
  int wid  = blockIdx.x * 8 + (threadIdx.x >> 5);
  int lane = threadIdx.x & 31;
  int b  = wid / (8 * 64);
  int r0 = wid % (8 * 64);
  int mT = r0 / 64;
  int nG = r0 % 64;
  int row = lane & 15, off = (lane >> 4) * 8;
  int m0 = mT * 16;

  v16h a[4];
  const _Float16* wrow = wp + (size_t)(m0 + row) * CH;
#pragma unroll
  for (int kk = 0; kk < 4; ++kk)
    a[kk] = cat8(ld8(wrow + kk * 32 + off), ld8(wrow + kk * 32 + off + 16));

  float bias[8];
#pragma unroll
  for (int r = 0; r < 8; ++r) bias[r] = pb[m0 + r + off];

  for (int nt = 0; nt < 4; ++nt) {
    int n = nG * 64 + nt * 16 + row;
    const _Float16* bc = aoT + ((size_t)b * HWN + n) * CH;
    v8f acc = {};
#pragma unroll
    for (int kk = 0; kk < 4; ++kk) {
      v16h bt = cat8(ld8(bc + kk * 32 + off), ld8(bc + kk * 32 + off + 16));
      acc = __builtin_amdgcn_wmma_f32_16x16x32_f16(false, a[kk], false, bt,
                                                   (short)0, acc, false, false);
    }
#pragma unroll
    for (int r = 0; r < 8; ++r) {
      size_t oi = ((size_t)b * CH + m0 + r + off) * HWN + n;
      out[oi] = acc[r] + bias[r] + x[oi];
    }
  }
}

extern "C" void kernel_launch(void* const* d_in, const int* in_sizes, int n_in,
                              void* d_out, int out_size, void* d_ws, size_t ws_size,
                              hipStream_t stream) {
  const float* x      = (const float*)d_in[0];
  const float* gn_w   = (const float*)d_in[1];
  const float* gn_b   = (const float*)d_in[2];
  const float* qkv_w  = (const float*)d_in[3];
  const float* qkv_b  = (const float*)d_in[4];
  const float* proj_w = (const float*)d_in[5];
  const float* proj_b = (const float*)d_in[6];
  float* out = (float*)d_out;

  char* ws = (char*)d_ws;
  const size_t SZ_XNT = (size_t)BATCH * HWN * CH * 2;      // 4 MB
  const size_t SZ_QKV = (size_t)BATCH * NH * HWN * DH * 2; // 4 MB each
  float*     stats = (float*)ws;                    size_t o = 256;
  _Float16*  xnt   = (_Float16*)(ws + o);           o += SZ_XNT;
  _Float16*  wq    = (_Float16*)(ws + o);           o += (size_t)TC * CH * 2;
  _Float16*  wp    = (_Float16*)(ws + o);           o += (size_t)CH * CH * 2;
  _Float16*  qT    = (_Float16*)(ws + o);           o += SZ_QKV;
  _Float16*  kT    = (_Float16*)(ws + o);           o += SZ_QKV;
  _Float16*  vD    = (_Float16*)(ws + o);           o += SZ_QKV;
  _Float16*  aoT   = (_Float16*)(ws + o);           o += SZ_XNT;
  (void)ws_size; (void)in_sizes; (void)n_in; (void)out_size;

  stats_kernel<<<BATCH, 256, 0, stream>>>(x, stats);
  norm_kernel<<<(BATCH * CH * HWN) / 1024, 256, 0, stream>>>(x, gn_w, gn_b, stats, xnt);
  f2h_kernel<<<(TC * CH + 255) / 256, 256, 0, stream>>>(qkv_w, wq, TC * CH);
  f2h_kernel<<<(CH * CH + 255) / 256, 256, 0, stream>>>(proj_w, wp, CH * CH);
  qkv_gemm<<<(BATCH * 24 * 64) / 8, 256, 0, stream>>>(wq, xnt, qkv_b, qT, kT, vD);
  flash_attn<<<BATCH * NH * 32, 256, 0, stream>>>(qT, kT, vD, aoT);
  proj_gemm<<<(BATCH * 8 * 64) / 8, 256, 0, stream>>>(wp, aoT, proj_b, x, out);
}
